// KGramMLPSeqModel_86354612453704
// MI455X (gfx1250) — matrix operand order repaired and checked by reading, not compiled, not run on hardware
//
#include <hip/hip_runtime.h>
#include <hip/hip_bf16.h>

// ---------------------------------------------------------------------------
// KGram MLP for MI455X (gfx1250, wave32).
// Strategy: bf16 WMMA (v_wmma_f32_16x16x32_bf16) for all three GEMMs,
// f32 accumulate + f32 bias/SiLU epilogue. Dominant cost is L3
// (4096x1024x32000, 268 GFLOP) plus the 0.52 GB f32 logits write
// (~22us at 23.3 TB/s HBM). Weights are converted+transposed to bf16 once
// per call (W3^T bf16 = 65 MB -> resident in 192 MB L2 across the 64
// M-tile passes).
// ---------------------------------------------------------------------------

typedef __attribute__((ext_vector_type(16))) __bf16 v16bf;
typedef __attribute__((ext_vector_type(8)))  float  v8f;

#define KCTX   3
#define EMBED  1024
#define VOCAB  32000
#define SEQ    2048
#define BATCH  2
#define MROWS  (SEQ * BATCH)      // 4096
#define KIN    (KCTX * EMBED)     // 3072

// ---------------------------------------------------------------------------
// Transpose + f32->bf16 convert:  W (Kd x N, row-major f32) -> Wt (N x Kd bf16)
// 32x32 LDS tile, 256 threads.
// ---------------------------------------------------------------------------
__global__ void __launch_bounds__(256)
convert_wt_kernel(const float* __restrict__ W, __bf16* __restrict__ Wt,
                  int Kd, int N) {
    __shared__ float tile[32][33];          // +1 pad: no LDS bank conflicts
    const int tx = threadIdx.x & 31;
    const int ty = threadIdx.x >> 5;        // 0..7
    const int k0 = blockIdx.x * 32;
    const int n0 = blockIdx.y * 32;
    #pragma unroll
    for (int r = ty; r < 32; r += 8)
        tile[r][tx] = W[(size_t)(k0 + r) * N + (n0 + tx)];
    __syncthreads();
    #pragma unroll
    for (int r = ty; r < 32; r += 8)
        Wt[(size_t)(n0 + r) * Kd + (k0 + tx)] = (__bf16)tile[tx][r];
}

// ---------------------------------------------------------------------------
// Embedding gather: x[(s*B+b)*3072 + j*1024 + e] = emb[ctx(s,j,b)*1024 + e]
// ctx(s,j,b) = (s+j < K) ? 0 : tokens[(s+j-K)*B + b]   (left pad w/ token 0)
// One thread per bf16 output element; consecutive threads -> consecutive e.
// ---------------------------------------------------------------------------
__global__ void __launch_bounds__(256)
gather_x_kernel(const int* __restrict__ tokens,
                const float* __restrict__ emb,
                __bf16* __restrict__ xb) {
    const int idx = blockIdx.x * 256 + threadIdx.x;   // < 4096*3072
    const int e   = idx & (EMBED - 1);
    const int rj  = idx >> 10;          // r*3 + j
    const int j   = rj % KCTX;
    const int r   = rj / KCTX;          // s*BATCH + b
    const int b   = r & (BATCH - 1);
    const int s   = r >> 1;
    const int t   = s + j;
    const int tok = (t < KCTX) ? 0 : tokens[(t - KCTX) * BATCH + b];
    xb[idx] = (__bf16)emb[(size_t)tok * EMBED + e];
}

// ---------------------------------------------------------------------------
// WMMA operand loaders (wave32; this code's contiguous k order, not the ISA table order; A and B match):
//  A 16x32 bf16: lane L holds row M=L&15; lanes>=16 have K-offset +16;
//                elems 0..15 = 16 contiguous K values.
//  B 32x16 bf16: lane L holds col N=L&15; lanes>=16 have K-offset +16;
//                elems 0..15 = 16 contiguous K values.
// ---------------------------------------------------------------------------
__device__ __forceinline__ v16bf load_a16x32(const __bf16* p) {
    v16bf r;
    *((uint4*)&r)     = *(const uint4*)(p);        // 8 bf16 @ +0
    *((uint4*)&r + 1) = *(const uint4*)(p + 8);    // 8 bf16 @ +8 elems
    return r;
}
__device__ __forceinline__ v16bf load_b32x16(const __bf16* p) {
    v16bf r;
    *((uint4*)&r)     = *(const uint4*)(p);        // 8 bf16 @ +0
    *((uint4*)&r + 1) = *(const uint4*)(p + 8);    // 8 bf16 @ +8 elems
    return r;
}

// ---------------------------------------------------------------------------
// GEMM:  C(MxN) = act( A(MxKd) @ Bt(NxKd)^T + bias )
// A row-major bf16, Bt row-major bf16 (pre-transposed weight).
// Block = 256 threads = 8 waves; waves tiled 2(M) x 4(N);
// wave tile = 32x64 -> block tile 64x256; 8 wmma per K=32 step.
// ---------------------------------------------------------------------------
template <bool SILU, bool OUT_BF16>
__global__ void __launch_bounds__(256)
gemm_bias_act_kernel(const __bf16* __restrict__ A,
                     const __bf16* __restrict__ Bt,
                     const float*  __restrict__ bias,
                     void* __restrict__ Cout,
                     int N, int Kd) {
    const int lane  = threadIdx.x & 31;
    const int wave  = threadIdx.x >> 5;       // 0..7
    const int row   = lane & 15;              // M-row / N-col / out-N per lane
    const int hi    = lane >> 4;              // lane group 0/1

    const int m0 = blockIdx.x * 64  + (wave & 1) * 32;
    const int n0 = blockIdx.y * 256 + (wave >> 1) * 64;

    v8f acc[2][4] = {};

    const __bf16* aB0 = A + (size_t)(m0 + row)      * Kd + hi * 16;
    const __bf16* aB1 = A + (size_t)(m0 + 16 + row) * Kd + hi * 16;
    const __bf16* bB0 = Bt + (size_t)(n0 +  0 + row) * Kd + hi * 16;
    const __bf16* bB1 = Bt + (size_t)(n0 + 16 + row) * Kd + hi * 16;
    const __bf16* bB2 = Bt + (size_t)(n0 + 32 + row) * Kd + hi * 16;
    const __bf16* bB3 = Bt + (size_t)(n0 + 48 + row) * Kd + hi * 16;

    for (int k = 0; k < Kd; k += 32) {
        __builtin_prefetch(aB0 + k + 128, 0, 1);   // global_prefetch_b8
        __builtin_prefetch(bB0 + k + 128, 0, 1);

        const v16bf a0 = load_a16x32(aB0 + k);
        const v16bf a1 = load_a16x32(aB1 + k);
        const v16bf b0 = load_b32x16(bB0 + k);
        const v16bf b1 = load_b32x16(bB1 + k);
        const v16bf b2 = load_b32x16(bB2 + k);
        const v16bf b3 = load_b32x16(bB3 + k);

        acc[0][0] = __builtin_amdgcn_wmma_f32_16x16x32_bf16(false, a0, false, b0, (short)0, acc[0][0], false, false);
        acc[0][1] = __builtin_amdgcn_wmma_f32_16x16x32_bf16(false, a0, false, b1, (short)0, acc[0][1], false, false);
        acc[0][2] = __builtin_amdgcn_wmma_f32_16x16x32_bf16(false, a0, false, b2, (short)0, acc[0][2], false, false);
        acc[0][3] = __builtin_amdgcn_wmma_f32_16x16x32_bf16(false, a0, false, b3, (short)0, acc[0][3], false, false);
        acc[1][0] = __builtin_amdgcn_wmma_f32_16x16x32_bf16(false, a1, false, b0, (short)0, acc[1][0], false, false);
        acc[1][1] = __builtin_amdgcn_wmma_f32_16x16x32_bf16(false, a1, false, b1, (short)0, acc[1][1], false, false);
        acc[1][2] = __builtin_amdgcn_wmma_f32_16x16x32_bf16(false, a1, false, b2, (short)0, acc[1][2], false, false);
        acc[1][3] = __builtin_amdgcn_wmma_f32_16x16x32_bf16(false, a1, false, b3, (short)0, acc[1][3], false, false);
    }

    // Epilogue: C/D layout -> element (m0+i*16 + v + 8*hi, n0+j*16 + row)
    #pragma unroll
    for (int j = 0; j < 4; ++j) {
        const int n  = n0 + j * 16 + row;
        const float bj = bias[n];
        #pragma unroll
        for (int i = 0; i < 2; ++i) {
            #pragma unroll
            for (int v = 0; v < 8; ++v) {
                float x = acc[i][j][v] + bj;
                if (SILU) x = x / (1.0f + __expf(-x));
                const int m = m0 + i * 16 + v + hi * 8;
                if (OUT_BF16)
                    ((__bf16*)Cout)[(size_t)m * N + n] = (__bf16)x;
                else
                    ((float*)Cout)[(size_t)m * N + n] = x;
            }
        }
    }
}

// ---------------------------------------------------------------------------
// Launch
// ---------------------------------------------------------------------------
extern "C" void kernel_launch(void* const* d_in, const int* in_sizes, int n_in,
                              void* d_out, int out_size, void* d_ws, size_t ws_size,
                              hipStream_t stream) {
    const int*   tokens = (const int*)  d_in[0];   // (2048, 2)
    const float* emb    = (const float*)d_in[1];   // (32000, 1024)
    const float* W1     = (const float*)d_in[2];   // (3072, 1024)
    const float* b1     = (const float*)d_in[3];   // (1024,)
    const float* W2     = (const float*)d_in[4];   // (1024, 1024)
    const float* b2     = (const float*)d_in[5];   // (1024,)
    const float* W3     = (const float*)d_in[6];   // (1024, 32000)
    const float* b3     = (const float*)d_in[7];   // (32000,)
    float* out = (float*)d_out;                    // (2048, 2, 32000) f32

    // Workspace layout (bytes, all 256-aligned)
    char* ws = (char*)d_ws;
    __bf16* xb  = (__bf16*)(ws);                                  // 4096x3072  bf16 (25.2 MB)
    __bf16* w1t = (__bf16*)(ws + (size_t)25165824);               // 1024x3072  bf16 ( 6.3 MB)
    __bf16* w2t = (__bf16*)(ws + (size_t)31457280);               // 1024x1024  bf16 ( 2.1 MB)
    __bf16* w3t = (__bf16*)(ws + (size_t)33554432);               // 32000x1024 bf16 (65.5 MB)
    __bf16* h1  = (__bf16*)(ws + (size_t)99090432);               // 4096x1024  bf16 ( 8.4 MB)
    __bf16* h2  = (__bf16*)(ws + (size_t)107479040);              // 4096x1024  bf16 ( 8.4 MB)

    // 1) Weight transpose + bf16 convert
    convert_wt_kernel<<<dim3(KIN / 32,   EMBED / 32), 256, 0, stream>>>(W1, w1t, KIN,   EMBED);
    convert_wt_kernel<<<dim3(EMBED / 32, EMBED / 32), 256, 0, stream>>>(W2, w2t, EMBED, EMBED);
    convert_wt_kernel<<<dim3(EMBED / 32, VOCAB / 32), 256, 0, stream>>>(W3, w3t, EMBED, VOCAB);

    // 2) Embedding gather -> x (bf16)
    gather_x_kernel<<<(MROWS * KIN) / 256, 256, 0, stream>>>(tokens, emb, xb);

    // 3) L1: h1 = silu(x @ W1 + b1)          (4096x3072)@(3072x1024)
    gemm_bias_act_kernel<true, true><<<dim3(MROWS / 64, EMBED / 256), 256, 0, stream>>>(
        xb, w1t, b1, (void*)h1, EMBED, KIN);

    // 4) L2: h2 = silu(h1 @ W2 + b2)         (4096x1024)@(1024x1024)
    gemm_bias_act_kernel<true, true><<<dim3(MROWS / 64, EMBED / 256), 256, 0, stream>>>(
        h1, w2t, b2, (void*)h2, EMBED, EMBED);

    // 5) L3: out = h2 @ W3 + b3              (4096x1024)@(1024x32000), f32 out
    gemm_bias_act_kernel<false, false><<<dim3(MROWS / 64, VOCAB / 256), 256, 0, stream>>>(
        h2, w3t, b3, (void*)out, VOCAB, EMBED);

    (void)in_sizes; (void)n_in; (void)out_size; (void)ws_size;
}
